// SparseLinear_3032246911256
// MI455X (gfx1250) — matrix operand order repaired and checked by reading, not compiled, not run on hardware
//
#include <hip/hip_runtime.h>
#include <stdint.h>

typedef __attribute__((ext_vector_type(16))) _Float16 v16h;
typedef __attribute__((ext_vector_type(8)))  float    v8f;

#define IN_F   2048
#define OUT_F  2048
#define NNZ_   262144
#define NNODES 65536

// GEMM tiling
#define BM 128
#define BO 128
#define BK 32
#define LDA 40    // padded A-tile stride (f16 elems) -> 80B rows, 16B aligned
#define LDB 136   // padded B-tile stride (f16 elems) -> 272B rows, 16B aligned

// ---------------- weight build ----------------

__global__ void zero_w_kernel(float* __restrict__ w) {
  size_t i = (size_t)blockIdx.x * blockDim.x + threadIdx.x;
  const size_t n = (size_t)IN_F * OUT_F;
  const size_t step = (size_t)gridDim.x * blockDim.x;
  for (; i < n; i += step) w[i] = 0.0f;
}

// Wt[i][o] = sum of values at (row=o, col=i)  (duplicate coords accumulate)
__global__ void scatter_w_kernel(const float* __restrict__ vals,
                                 const int* __restrict__ rows,
                                 const int* __restrict__ cols,
                                 float* __restrict__ wt) {
  int e = blockIdx.x * blockDim.x + threadIdx.x;
  if (e < NNZ_) {
    atomicAdd(&wt[(size_t)cols[e] * OUT_F + rows[e]], vals[e]);
  }
}

__global__ void convert_w_kernel(const float* __restrict__ wt,
                                 _Float16* __restrict__ wt16) {
  size_t i = (size_t)blockIdx.x * blockDim.x + threadIdx.x;
  const size_t n = (size_t)IN_F * OUT_F;
  const size_t step = (size_t)gridDim.x * blockDim.x;
  for (; i < n; i += step) wt16[i] = (_Float16)wt[i];
}

// ---------------- WMMA GEMM: y = x @ Wt + bias ----------------
// x:[N,K] f32 (converted to f16 while staging), Wt:[K,O] f16, y:[N,O] f32.
// Block = 128x128 output tile, 256 threads = 8 wave32; wave (wm,wo) owns a
// 64x32 subtile = 4x2 v_wmma_f32_16x16x32_f16 accumulators.
// Double-buffered LDS; B tile staged with global_load_async_to_lds_b128
// (ASYNCcnt), A tile staged through VGPRs (needs f32->f16 convert).

__global__ __launch_bounds__(256) void gemm_wmma_kernel(
    const float*    __restrict__ x,
    const _Float16* __restrict__ wt,
    const float*    __restrict__ bias,
    float*          __restrict__ y) {

  __shared__ __align__(16) _Float16 la[2][BM * LDA];
  __shared__ __align__(16) _Float16 lb[2][BK * LDB];

  const int t    = threadIdx.x;
  const int lane = t & 31;
  const int w    = t >> 5;
  const int wm   = w & 1;   // 0..1 : 64-row slabs
  const int wo   = w >> 1;  // 0..3 : 32-col slabs
  const int lrow = lane & 15;
  const int half = lane >> 4;

  const int n0 = blockIdx.y * BM;
  const int o0 = blockIdx.x * BO;

  // init accumulators with bias (C layout: VGPR r -> M=r+8*half, N=lrow)
  v8f acc[4][2];
#pragma unroll
  for (int o = 0; o < 2; ++o) {
    const float b = bias[o0 + wo * 32 + o * 16 + lrow];
    v8f c;
#pragma unroll
    for (int r = 0; r < 8; ++r) c[r] = b;
#pragma unroll
    for (int m = 0; m < 4; ++m) acc[m][o] = c;
  }

  // staging assignments
  const int arow = t >> 1;         // 0..127
  const int aseg = t & 1;          // two 16-elem segments per row
  const int brow = t >> 3;         // 0..31 (K)
  const int bcol = (t & 7) << 4;   // 8 x 16-elem segments per K row

  const float*    xsrc = x  + (size_t)(n0 + arow) * IN_F + aseg * 16;
  const _Float16* wsrc = wt + (size_t)brow * OUT_F + o0 + bcol;

  auto stage = [&](int p, int kt) {
    // ---- A tile: x[n0+arow][kt + aseg*16 .. +16), f32 -> f16, ds_store ----
    {
      const float4* src = (const float4*)(xsrc + kt);
      float4 f[4];
#pragma unroll
      for (int i = 0; i < 4; ++i) f[i] = src[i];
      union { _Float16 h[16]; uint4 u[2]; } pk;
      const float* ff = (const float*)f;
#pragma unroll
      for (int i = 0; i < 16; ++i) pk.h[i] = (_Float16)ff[i];
      uint4* dst = (uint4*)&la[p][arow * LDA + aseg * 16];
      dst[0] = pk.u[0];
      dst[1] = pk.u[1];
    }
    // ---- B tile: wt[kt+brow][o0+bcol .. +16), async global -> LDS ----
    {
      const _Float16* gp = wsrc + (size_t)kt * OUT_F;   // 32B per thread
      uint64_t ga = (uint64_t)(uintptr_t)gp;
      uint32_t ld = (uint32_t)(uintptr_t)&lb[p][brow * LDB + bcol];
      // INST_OFFSET applies to both the global and the LDS address.
      asm volatile(
          "global_load_async_to_lds_b128 %0, %1, off\n\t"
          "global_load_async_to_lds_b128 %0, %1, off offset:16"
          :: "v"(ld), "v"(ga)
          : "memory");
    }
  };

  // prologue: stage first tile
  stage(0, 0);
  asm volatile("s_wait_asynccnt 0" ::: "memory");
  __syncthreads();

  int p = 0;
  for (int kt = 0; kt < IN_F; kt += BK) {
    // stage next tile into the other buffer while computing this one
    if (kt + BK < IN_F) stage(p ^ 1, kt + BK);

    // ---- A fragments: lane(l<16) K={0..7,16..23}, lane(l>=16) K={8..15,24..31}
    v16h af[4];
#pragma unroll
    for (int m = 0; m < 4; ++m) {
      const _Float16* q = &la[p][(wm * 64 + m * 16 + lrow) * LDA + half * 8];
      union { uint4 u[2]; v16h v; } u;
      u.u[0] = *(const uint4*)(q);
      u.u[1] = *(const uint4*)(q + 16);
      af[m] = u.v;
    }
    // ---- B fragments: column N = lrow, K = {8h..8h+7, 16+8h..16+8h+7} per lane
    v16h bf[2];
#pragma unroll
    for (int o = 0; o < 2; ++o) {
      const _Float16* q = &lb[p][wo * 32 + o * 16 + lrow];
#pragma unroll
      for (int s = 0; s < 16; ++s) {
        const int k = (s < 8) ? (half * 8 + s) : (16 + half * 8 + (s - 8));
        bf[o][s] = q[k * LDB];
      }
    }

#pragma unroll
    for (int m = 0; m < 4; ++m)
#pragma unroll
      for (int o = 0; o < 2; ++o)
        acc[m][o] = __builtin_amdgcn_wmma_f32_16x16x32_f16(
            false, af[m], false, bf[o], (short)0, acc[m][o], false, false);

    // own async B-writes for tile kt+BK must be in LDS before the joint
    // barrier; __syncthreads adds the dscnt wait for the A ds_stores.
    asm volatile("s_wait_asynccnt 0" ::: "memory");
    __syncthreads();
    p ^= 1;
  }

  // ---- epilogue: C layout VGPR r -> (M = r + 8*half, N = lrow) ----
#pragma unroll
  for (int m = 0; m < 4; ++m) {
#pragma unroll
    for (int o = 0; o < 2; ++o) {
      const int col = o0 + wo * 32 + o * 16 + lrow;
#pragma unroll
      for (int r = 0; r < 8; ++r) {
        const int row = n0 + wm * 64 + m * 16 + half * 8 + r;
        y[(size_t)row * OUT_F + col] = acc[m][o][r];
      }
    }
  }
}

// ---------------- launch ----------------

extern "C" void kernel_launch(void* const* d_in, const int* in_sizes, int n_in,
                              void* d_out, int out_size, void* d_ws, size_t ws_size,
                              hipStream_t stream) {
  const float* x    = (const float*)d_in[0];
  const float* vals = (const float*)d_in[1];
  const float* bias = (const float*)d_in[2];
  const int*   rows = (const int*)d_in[3];
  const int*   cols = (const int*)d_in[4];
  float*       y    = (float*)d_out;

  float*    wacc = (float*)d_ws;                              // 16 MB f32 Wt
  _Float16* wt16 = (_Float16*)((char*)d_ws +
                               (size_t)IN_F * OUT_F * sizeof(float)); // 8 MB f16 Wt

  zero_w_kernel<<<2048, 256, 0, stream>>>(wacc);
  scatter_w_kernel<<<NNZ_ / 256, 256, 0, stream>>>(vals, rows, cols, wacc);
  convert_w_kernel<<<2048, 256, 0, stream>>>(wacc, wt16);

  dim3 grid(OUT_F / BO, NNODES / BM);
  gemm_wmma_kernel<<<grid, 256, 0, stream>>>(x, wt16, bias, y);
}
